// MultiHeadAttentionLayer_65274912964954
// MI455X (gfx1250) — matrix operand order repaired and checked by reading, not compiled, not run on hardware
//
#include <hip/hip_runtime.h>
#include <hip/hip_bf16.h>
#include <math.h>

// ---------- types ----------
typedef __attribute__((ext_vector_type(16))) _Float16 v16h;
typedef __attribute__((ext_vector_type(8)))  _Float16 v8h;
typedef __attribute__((ext_vector_type(8)))  float    v8f;
typedef __attribute__((ext_vector_type(4)))  float    v4f;

#define B_   8
#define S_   1024
#define H_   16
#define HD_  64
#define D_   1024
#define BH_  (B_ * H_)
#define M_   (B_ * S_)     // 8192 rows for projections

static __device__ __forceinline__ v16h cat16(v8h lo, v8h hi) {
  return __builtin_shufflevector(lo, hi, 0,1,2,3,4,5,6,7,8,9,10,11,12,13,14,15);
}

// two v4f -> packed 8 x f16 (lowers to v_cvt_pk_f16_f32 x4)
static __device__ __forceinline__ v8h pack8(v4f a, v4f b) {
  v8h r;
  r[0] = (_Float16)a[0]; r[1] = (_Float16)a[1];
  r[2] = (_Float16)a[2]; r[3] = (_Float16)a[3];
  r[4] = (_Float16)b[0]; r[5] = (_Float16)b[1];
  r[6] = (_Float16)b[2]; r[7] = (_Float16)b[3];
  return r;
}

static __device__ __forceinline__ v8h cvt8(const float* __restrict__ p) {
  return pack8(*(const v4f*)p, *(const v4f*)(p + 4));
}

static __device__ __forceinline__ v16h cvt16(const float* __restrict__ p) {
  return cat16(cvt8(p), cvt8(p + 8));
}

static __device__ __forceinline__ v8f wmma16(v16h a, v16h b, v8f c) {
  // D = A(16x32 f16) * B(32x16 f16) + C(16x16 f32)
  return __builtin_amdgcn_wmma_f32_16x16x32_f16(false, a, false, b,
                                                (short)0, c, false, false);
}

// =====================================================================
// Kernel 1/5: Y = A[M,K] @ W[N,K]^T + bias, f16 WMMA with f32 accum.
// Block tile 64(M) x 128(N), 256 threads = 8 waves as 2(M) x 4(N),
// each wave 32x32 = 2x2 WMMA fragments.  Double-buffered LDS, software
// pipelined: next tile's global_load_b128s issue before current tile's
// WMMAs, so the loadcnt wait is covered by compute.  One barrier/iter.
// =====================================================================
#define LDSK 40   // 32 k-halves + 8 pad (keeps 16B alignment, skews banks)

__global__ __launch_bounds__(256)
void mha_gemm_bias(const float* __restrict__ A, const float* __restrict__ W,
                   const float* __restrict__ bias, float* __restrict__ Y,
                   int K, int N, int split_heads) {
  __shared__ _Float16 Xs[2][64  * LDSK];
  __shared__ _Float16 Ws[2][128 * LDSK];

  const int tid  = threadIdx.x;
  const int lane = tid & 31;
  const int wave = tid >> 5;
  const int wm   = wave & 1;        // 2 waves along M
  const int wn   = wave >> 1;       // 4 waves along N
  const int bm   = blockIdx.x * 64;
  const int bn   = blockIdx.y * 128;
  const int lr   = lane & 15;
  const int hf   = lane >> 4;

  // staging coordinates: 4 threads per 32-wide row, 8 floats each
  const int sr = tid >> 2;          // 0..63
  const int sc = (tid & 3) * 8;     // 0,8,16,24
  const float* aSrc  = A + (size_t)(bm + sr) * K + sc;
  const float* wSrc0 = W + (size_t)(bn + sr) * K + sc;
  const float* wSrc1 = W + (size_t)(bn + sr + 64) * K + sc;

  v4f ra0, ra1, rw00, rw01, rw10, rw11;
  auto load_tile = [&](int k0) {
    ra0  = *(const v4f*)(aSrc  + k0); ra1  = *(const v4f*)(aSrc  + k0 + 4);
    rw00 = *(const v4f*)(wSrc0 + k0); rw01 = *(const v4f*)(wSrc0 + k0 + 4);
    rw10 = *(const v4f*)(wSrc1 + k0); rw11 = *(const v4f*)(wSrc1 + k0 + 4);
  };
  auto store_tile = [&](int buf) {
    *(v8h*)&Xs[buf][sr * LDSK + sc]        = pack8(ra0, ra1);
    *(v8h*)&Ws[buf][sr * LDSK + sc]        = pack8(rw00, rw01);
    *(v8h*)&Ws[buf][(sr + 64) * LDSK + sc] = pack8(rw10, rw11);
  };

  v8f acc[2][2] = {{{0.f}, {0.f}}, {{0.f}, {0.f}}};

  load_tile(0);
  store_tile(0);
  __syncthreads();

  int cur = 0;
  for (int k0 = 0; k0 < K; k0 += 32) {
    const bool more = (k0 + 32) < K;
    if (more) load_tile(k0 + 32);       // issue next tile early

    v16h afrag[2], bfrag[2];
#pragma unroll
    for (int fm = 0; fm < 2; ++fm) {
      const _Float16* base = &Xs[cur][(wm * 32 + fm * 16 + lr) * LDSK];
      afrag[fm] = cat16(*(const v8h*)(base + 16 * hf),
                        *(const v8h*)(base + 16 * hf + 8));
    }
#pragma unroll
    for (int fn = 0; fn < 2; ++fn) {
      const _Float16* base = &Ws[cur][(wn * 32 + fn * 16 + lr) * LDSK + 16 * hf];
      bfrag[fn] = cat16(*(const v8h*)(base), *(const v8h*)(base + 8));
    }
#pragma unroll
    for (int fm = 0; fm < 2; ++fm)
#pragma unroll
      for (int fn = 0; fn < 2; ++fn)
        acc[fm][fn] = wmma16(afrag[fm], bfrag[fn], acc[fm][fn]);

    if (more) {
      store_tile(cur ^ 1);              // fill the other buffer
      __syncthreads();
    }
    cur ^= 1;
  }

#pragma unroll
  for (int fm = 0; fm < 2; ++fm)
#pragma unroll
    for (int fn = 0; fn < 2; ++fn)
#pragma unroll
      for (int r = 0; r < 8; ++r) {
        int row = bm + wm * 32 + fm * 16 + r + 8 * hf;
        int col = bn + wn * 32 + fn * 16 + lr;
        float v = acc[fm][fn][r] + bias[col];
        if (split_heads) {  // [B,H,S,Hd]
          int b = row >> 10, s = row & 1023;
          int h = col >> 6,  d = col & 63;
          Y[(((size_t)b * H_ + h) * S_ + s) * HD_ + d] = v;
        } else {
          Y[(size_t)row * N + col] = v;
        }
      }
}

// =====================================================================
// Kernel 2: raw scaled scores = Q K^T / 8.  One block = (bh, 16 q-rows),
// 8 waves each own a 128-key strip (8 fragments), K-dim = 64 = 2 steps.
// Q/K tiles are L2-resident; fragments converted f32->f16 on the fly.
// =====================================================================
__global__ __launch_bounds__(256)
void mha_scores(const float* __restrict__ Q, const float* __restrict__ Km,
                float* __restrict__ attn) {
  const int bh   = blockIdx.y;
  const int q0   = blockIdx.x * 16;
  const int lane = threadIdx.x & 31;
  const int wave = threadIdx.x >> 5;
  const int lr   = lane & 15;
  const int hf   = lane >> 4;

  const float* qb = Q  + (size_t)bh * S_ * HD_;
  const float* kb = Km + (size_t)bh * S_ * HD_;

  v8f acc[8] = {{0.f}};
#pragma unroll
  for (int k0 = 0; k0 < 64; k0 += 32) {
    const float* qrow = qb + (size_t)(q0 + lr) * HD_;
    v16h a = cat16(cvt8(qrow + k0 + 16 * hf),
                   cvt8(qrow + k0 + 16 * hf + 8));
#pragma unroll
    for (int f = 0; f < 8; ++f) {
      int n = wave * 128 + f * 16 + lr;
      v16h b = cvt16(kb + (size_t)n * HD_ + k0 + 16 * hf);
      acc[f] = wmma16(a, b, acc[f]);
    }
  }

  const float scale = 0.125f;  // 1/sqrt(64)
  float* ob = attn + (size_t)bh * S_ * S_;
#pragma unroll
  for (int f = 0; f < 8; ++f)
#pragma unroll
    for (int r = 0; r < 8; ++r) {
      int row = q0 + r + 8 * hf;
      int col = wave * 128 + f * 16 + lr;
      ob[(size_t)row * S_ + col] = acc[f][r] * scale;
    }
}

// =====================================================================
// Kernel 3: in-place row softmax over 1024 columns. One row per block.
// Each thread owns 4 contiguous floats: b128 in, b128 out.
// =====================================================================
__global__ __launch_bounds__(256)
void mha_softmax(float* __restrict__ attn) {
  float* p = attn + (size_t)blockIdx.x * S_ + threadIdx.x * 4;
  const int tid = threadIdx.x;
  __shared__ float red[256];

  v4f v = *(const v4f*)p;
  float m = fmaxf(fmaxf(v[0], v[1]), fmaxf(v[2], v[3]));
  red[tid] = m; __syncthreads();
  for (int s = 128; s > 0; s >>= 1) {
    if (tid < s) red[tid] = fmaxf(red[tid], red[tid + s]);
    __syncthreads();
  }
  m = red[0]; __syncthreads();

  v[0] = __expf(v[0] - m); v[1] = __expf(v[1] - m);
  v[2] = __expf(v[2] - m); v[3] = __expf(v[3] - m);
  red[tid] = (v[0] + v[1]) + (v[2] + v[3]); __syncthreads();
  for (int s = 128; s > 0; s >>= 1) {
    if (tid < s) red[tid] += red[tid + s];
    __syncthreads();
  }
  const float inv = 1.f / red[0];
  v[0] *= inv; v[1] *= inv; v[2] *= inv; v[3] *= inv;
  *(v4f*)p = v;
}

// =====================================================================
// Kernel 4: ctx = attn @ V per head.  Block = (bh, 64 q-rows), N = 64,
// K = 1024 keys in 32-steps.  V tile staged *transposed* in LDS so each
// B-fragment is a contiguous run of K.  Double-buffered + pipelined.
// Output written merged [B,S,D].
// =====================================================================
__global__ __launch_bounds__(256)
void mha_ctx(const float* __restrict__ attn, const float* __restrict__ V,
             float* __restrict__ ctx) {
  __shared__ _Float16 As[2][64 * LDSK];
  __shared__ _Float16 Vt[2][64 * LDSK];   // Vt[n][kk]

  const int bh   = blockIdx.y;
  const int m0   = blockIdx.x * 64;
  const int tid  = threadIdx.x;
  const int lane = tid & 31;
  const int wave = tid >> 5;
  const int wm   = wave & 3;          // 4 waves along M (16 rows each)
  const int wn   = wave >> 2;         // 2 waves along N (32 cols each)
  const int lr   = lane & 15;
  const int hf   = lane >> 4;

  const float* ab = attn + (size_t)bh * S_ * S_;
  const float* vb = V    + (size_t)bh * S_ * HD_;

  // attn staging: 4 threads per 32-wide row, 8 floats each
  const int sr = tid >> 2;            // 0..63
  const int sc = (tid & 3) * 8;       // 0,8,16,24
  const float* aSrc = ab + (size_t)(m0 + sr) * S_ + sc;
  // V staging (transposed): 8 threads per 64-wide key row, 8 n's each
  const int vk = tid >> 3;            // 0..31 (key within tile)
  const int vn = (tid & 7) * 8;       // n chunk
  const float* vSrc = vb + (size_t)vk * HD_ + vn;

  v4f ra0, ra1, rv0, rv1;
  auto load_tile = [&](int k0) {
    ra0 = *(const v4f*)(aSrc + k0);
    ra1 = *(const v4f*)(aSrc + k0 + 4);
    rv0 = *(const v4f*)(vSrc + (size_t)k0 * HD_);
    rv1 = *(const v4f*)(vSrc + (size_t)k0 * HD_ + 4);
  };
  auto store_tile = [&](int buf) {
    *(v8h*)&As[buf][sr * LDSK + sc] = pack8(ra0, ra1);
    v8h hv = pack8(rv0, rv1);
#pragma unroll
    for (int j = 0; j < 8; ++j)
      Vt[buf][(vn + j) * LDSK + vk] = hv[j];
  };

  v8f acc[2] = {{0.f}, {0.f}};

  load_tile(0);
  store_tile(0);
  __syncthreads();

  int cur = 0;
  for (int k0 = 0; k0 < S_; k0 += 32) {
    const bool more = (k0 + 32) < S_;
    if (more) load_tile(k0 + 32);

    const _Float16* abase = &As[cur][(wm * 16 + lr) * LDSK];
    v16h a = cat16(*(const v8h*)(abase + 16 * hf),
                   *(const v8h*)(abase + 16 * hf + 8));
#pragma unroll
    for (int fn = 0; fn < 2; ++fn) {
      const _Float16* bbase = &Vt[cur][(wn * 32 + fn * 16 + lr) * LDSK + 16 * hf];
      v16h b = cat16(*(const v8h*)(bbase), *(const v8h*)(bbase + 8));
      acc[fn] = wmma16(a, b, acc[fn]);
    }

    if (more) {
      store_tile(cur ^ 1);
      __syncthreads();
    }
    cur ^= 1;
  }

  const int h = bh & (H_ - 1), bb = bh >> 4;
#pragma unroll
  for (int fn = 0; fn < 2; ++fn)
#pragma unroll
    for (int r = 0; r < 8; ++r) {
      int s = m0 + wm * 16 + r + 8 * hf;
      int d = wn * 32 + fn * 16 + lr;
      ctx[(((size_t)bb * S_ + s) * H_ + h) * HD_ + d] = acc[fn][r];
    }
}

// =====================================================================
extern "C" void kernel_launch(void* const* d_in, const int* in_sizes, int n_in,
                              void* d_out, int out_size, void* d_ws, size_t ws_size,
                              hipStream_t stream) {
  (void)in_sizes; (void)n_in; (void)out_size; (void)ws_size;

  const float* x  = (const float*)d_in[0];
  const float* Wq = (const float*)d_in[1]; const float* bq = (const float*)d_in[2];
  const float* Wk = (const float*)d_in[3]; const float* bk = (const float*)d_in[4];
  const float* Wv = (const float*)d_in[5]; const float* bv = (const float*)d_in[6];
  const float* Wo = (const float*)d_in[7]; const float* bo = (const float*)d_in[8];

  float* out  = (float*)d_out;                              // [8,1024,1024]
  float* attn = out + (size_t)B_ * S_ * D_;                 // [8,16,1024,1024]

  const size_t mat = (size_t)M_ * D_;                       // 8192*1024 floats
  float* q_ws   = (float*)d_ws;                             // [BH,S,64]
  float* k_ws   = q_ws + mat;
  float* v_ws   = k_ws + mat;
  float* ctx_ws = q_ws;   // Q is dead after mha_scores; alias to save 32MB

  dim3 blk(256);
  dim3 gproj(M_ / 64, D_ / 128);          // 128 x 8

  // 1) Q/K/V projections, written head-split
  mha_gemm_bias<<<gproj, blk, 0, stream>>>(x, Wq, bq, q_ws, D_, D_, 1);
  mha_gemm_bias<<<gproj, blk, 0, stream>>>(x, Wk, bk, k_ws, D_, D_, 1);
  mha_gemm_bias<<<gproj, blk, 0, stream>>>(x, Wv, bv, v_ws, D_, D_, 1);

  // 2) raw scores -> attn region
  mha_scores<<<dim3(S_ / 16, BH_), blk, 0, stream>>>(q_ws, k_ws, attn);

  // 3) in-place softmax (attn is a required output)
  mha_softmax<<<dim3(BH_ * S_), blk, 0, stream>>>(attn);

  // 4) ctx = attn @ V, written merged [B,S,D]
  mha_ctx<<<dim3(S_ / 64, BH_), blk, 0, stream>>>(attn, v_ws, ctx_ws);

  // 5) output projection
  mha_gemm_bias<<<gproj, blk, 0, stream>>>(ctx_ws, Wo, bo, out, D_, D_, 0);
}
